// Block_5162550690658
// MI455X (gfx1250) — matrix operand order repaired and checked by reading, not compiled, not run on hardware
//
#include <hip/hip_runtime.h>
#include <math.h>

typedef __attribute__((ext_vector_type(16))) _Float16 v16h;
typedef __attribute__((ext_vector_type(8)))  _Float16 v8h;
typedef __attribute__((ext_vector_type(8)))  float    v8f;

union U16 { v16h v; v8h h[2]; };

__device__ inline v8f wmma_f16(v16h a, v16h b, v8f c) {
  return __builtin_amdgcn_wmma_f32_16x16x32_f16(false, a, false, b, (short)0, c, false, false);
}

// ---- DPP16 butterfly reductions within each 16-lane half (pure VALU) ------
template <int CTRL>
__device__ inline float dpp_mov(float x) {
  int r = __builtin_amdgcn_update_dpp(0, __float_as_int(x), CTRL, 0xF, 0xF, true);
  return __int_as_float(r);
}
__device__ inline float red_max16(float x) {
  x = fmaxf(x, dpp_mov<0xB1>(x));    // quad_perm(1,0,3,2)  : xor 1
  x = fmaxf(x, dpp_mov<0x4E>(x));    // quad_perm(2,3,0,1)  : xor 2
  x = fmaxf(x, dpp_mov<0x141>(x));   // row_half_mirror     : xor 7
  x = fmaxf(x, dpp_mov<0x140>(x));   // row_mirror          : xor 15
  return x;
}
__device__ inline float red_add16(float x) {
  x += dpp_mov<0xB1>(x);
  x += dpp_mov<0x4E>(x);
  x += dpp_mov<0x141>(x);
  x += dpp_mov<0x140>(x);
  return x;
}

// ---------------------------------------------------------------------------
// f32 -> f16 conversion
// ---------------------------------------------------------------------------
__global__ void cvt_kernel(const float* __restrict__ s, _Float16* __restrict__ d, int n) {
  int i = blockIdx.x * blockDim.x + threadIdx.x;
  if (i < n) d[i] = (_Float16)s[i];
}

// ---------------------------------------------------------------------------
// LayerNorm: one block (256 threads) per row of 1024, output f16
// ---------------------------------------------------------------------------
__global__ __launch_bounds__(256) void ln_kernel(const float* __restrict__ x,
                                                 const float* __restrict__ w,
                                                 const float* __restrict__ b,
                                                 _Float16* __restrict__ out) {
  const int row = blockIdx.x;
  const int tid = threadIdx.x;
  const float* xr = x + (size_t)row * 1024;
  float4 v = ((const float4*)xr)[tid];
  float sum = v.x + v.y + v.z + v.w;
  float sq  = v.x * v.x + v.y * v.y + v.z * v.z + v.w * v.w;
  // reduce within wave: 16-lane DPP butterfly + one cross-half shuffle
  sum = red_add16(sum);
  sq  = red_add16(sq);
  sum += __shfl_xor(sum, 16, 32);
  sq  += __shfl_xor(sq, 16, 32);
  __shared__ float red[16];
  if ((tid & 31) == 0) { red[tid >> 5] = sum; red[8 + (tid >> 5)] = sq; }
  __syncthreads();
  float tot = 0.f, totq = 0.f;
#pragma unroll
  for (int i = 0; i < 8; ++i) { tot += red[i]; totq += red[8 + i]; }
  const float mu  = tot * (1.0f / 1024.0f);
  const float var = totq * (1.0f / 1024.0f) - mu * mu;
  const float rs  = rsqrtf(var + 1e-5f);
  const int base = tid * 4;
  float xs[4] = {v.x, v.y, v.z, v.w};
  _Float16* o = out + (size_t)row * 1024 + base;
#pragma unroll
  for (int e = 0; e < 4; ++e)
    o[e] = (_Float16)((xs[e] - mu) * rs * w[base + e] + b[base + e]);
}

// ---------------------------------------------------------------------------
// Generic WMMA GEMM: C[M,N] = A[M,K] * W[N,K]^T  (+bias, optional GELU/resid)
// 256 threads = 8 waves (2x4); wave tile = 64x64; block tile = 128x256.
// ---------------------------------------------------------------------------
__device__ inline float gelu_f(float v) {
  return 0.5f * v * (1.0f + tanhf(0.7978845608028654f * (v + 0.044715f * v * v * v)));
}

template <bool DOGELU, bool DORES, bool OUT16>
__global__ __launch_bounds__(256) void gemm_kernel(
    const _Float16* __restrict__ A, const _Float16* __restrict__ W,
    const float* __restrict__ bias, const float* __restrict__ resid,
    _Float16* __restrict__ o16, float* __restrict__ o32,
    int M, int N, int K) {
  const int lane = threadIdx.x & 31;
  const int wv   = threadIdx.x >> 5;
  const int lh   = lane >> 4;
  const int lnn  = lane & 15;
  const int rb = blockIdx.y * 128 + (wv >> 2) * 64;
  const int cb = blockIdx.x * 256 + (wv & 3) * 64;

  v8f acc[4][4];
#pragma unroll
  for (int i = 0; i < 4; ++i)
#pragma unroll
    for (int j = 0; j < 4; ++j)
#pragma unroll
      for (int r = 0; r < 8; ++r) acc[i][j][r] = 0.0f;

  const int koff = lh ? 16 : 0;
  const _Float16* arow[4];
  const _Float16* wrow[4];
#pragma unroll
  for (int i = 0; i < 4; ++i) arow[i] = A + (size_t)(rb + i * 16 + lnn) * K;
#pragma unroll
  for (int j = 0; j < 4; ++j) wrow[j] = W + (size_t)(cb + j * 16 + lnn) * K + 16 * lh;

  for (int kb = 0; kb < K; kb += 32) {
    U16 af[4], bf[4];
#pragma unroll
    for (int i = 0; i < 4; ++i) {
      af[i].h[0] = *(const v8h*)(arow[i] + kb + koff);
      af[i].h[1] = *(const v8h*)(arow[i] + kb + koff + 8);
    }
#pragma unroll
    for (int j = 0; j < 4; ++j) {
      bf[j].h[0] = *(const v8h*)(wrow[j] + kb);
      bf[j].h[1] = *(const v8h*)(wrow[j] + kb + 8);
    }
#pragma unroll
    for (int i = 0; i < 4; ++i)
#pragma unroll
      for (int j = 0; j < 4; ++j)
        acc[i][j] = wmma_f16(af[i].v, bf[j].v, acc[i][j]);
  }

#pragma unroll
  for (int j = 0; j < 4; ++j) {
    const int col = cb + j * 16 + lnn;
    const float bv = bias[col];
#pragma unroll
    for (int i = 0; i < 4; ++i) {
#pragma unroll
      for (int r = 0; r < 8; ++r) {
        const int row = rb + i * 16 + r + 8 * lh;
        float v = acc[i][j][r] + bv;
        if (DOGELU) v = gelu_f(v);
        if (DORES)  v += resid[(size_t)row * N + col];
        if (OUT16)  o16[(size_t)row * N + col] = (_Float16)v;
        else        o32[(size_t)row * N + col] = v;
      }
    }
  }
}

// ---------------------------------------------------------------------------
// Transpose V slice of qkv16 into vT[b][c][t]  (coalesced read over c)
// ---------------------------------------------------------------------------
__global__ void vtrans_kernel(const _Float16* __restrict__ qkv,
                              _Float16* __restrict__ vT) {
  int idx = blockIdx.x * blockDim.x + threadIdx.x;  // B*T*C = 8388608
  if (idx >= 4 * 2048 * 1024) return;
  const int c = idx & 1023;
  const int t = (idx >> 10) & 2047;
  const int b = idx >> 21;
  vT[((size_t)b * 1024 + c) * 2048 + t] =
      qkv[((size_t)b * 2048 + t) * 3072 + 2048 + c];
}

// ---------------------------------------------------------------------------
// Flash attention: 1 wave per (b, h, 16-query tile); 128 threads = 4 waves.
// ---------------------------------------------------------------------------
__global__ __launch_bounds__(128) void attn_kernel(
    const _Float16* __restrict__ qkv, const _Float16* __restrict__ vT,
    _Float16* __restrict__ yout) {
  __shared__ __align__(16) _Float16 pbuf[4][16][32];

  const int lane = threadIdx.x & 31;
  const int wv   = threadIdx.x >> 5;
  const int lh   = lane >> 4;
  const int lnn  = lane & 15;

  const int wid = blockIdx.x * 4 + wv;
  const int b   = wid >> 11;          // 2048 waves per batch (16 heads * 128 tiles)
  const int rem = wid & 2047;
  const int h   = rem >> 7;
  const int qt  = rem & 127;
  const int qbase = qt * 16;
  const size_t rowbase = (size_t)b * 2048;

  // Q fragments (A-layout): rows qbase+lnn, hd split per half-wave
  U16 qf[2];
  {
    const _Float16* qp = qkv + (rowbase + qbase + lnn) * 3072 + h * 64;
#pragma unroll
    for (int f = 0; f < 2; ++f) {
      qf[f].h[0] = *(const v8h*)(qp + 32 * f + 16 * lh);
      qf[f].h[1] = *(const v8h*)(qp + 32 * f + 16 * lh + 8);
    }
  }

  v8f o[4];
#pragma unroll
  for (int nc = 0; nc < 4; ++nc)
#pragma unroll
    for (int r = 0; r < 8; ++r) o[nc][r] = 0.0f;
  float mrow[8], lrow[8];
#pragma unroll
  for (int r = 0; r < 8; ++r) { mrow[r] = -1e30f; lrow[r] = 0.0f; }

  const int njt = (qbase + 47) >> 5;   // 32-key tiles covering [0, qbase+15]
  for (int jt = 0; jt < njt; ++jt) {
    const int k0 = jt * 32;

    // ---- S = Q K^T for two 16-key subtiles ----
    float s0[8], s1[8];
    {
      const _Float16* kp = qkv + (rowbase + k0 + lnn) * 3072 + 1024 + h * 64;
      v8f sa;
#pragma unroll
      for (int r = 0; r < 8; ++r) sa[r] = 0.0f;
      U16 kf;
      kf.h[0] = *(const v8h*)(kp + 16 * lh);
      kf.h[1] = *(const v8h*)(kp + 16 * lh + 8);
      sa = wmma_f16(qf[0].v, kf.v, sa);
      kf.h[0] = *(const v8h*)(kp + 32 + 16 * lh);
      kf.h[1] = *(const v8h*)(kp + 32 + 16 * lh + 8);
      sa = wmma_f16(qf[1].v, kf.v, sa);
#pragma unroll
      for (int r = 0; r < 8; ++r) s0[r] = sa[r] * 0.125f;
    }
    {
      const _Float16* kp = qkv + (rowbase + k0 + 16 + lnn) * 3072 + 1024 + h * 64;
      v8f sa;
#pragma unroll
      for (int r = 0; r < 8; ++r) sa[r] = 0.0f;
      U16 kf;
      kf.h[0] = *(const v8h*)(kp + 16 * lh);
      kf.h[1] = *(const v8h*)(kp + 16 * lh + 8);
      sa = wmma_f16(qf[0].v, kf.v, sa);
      kf.h[0] = *(const v8h*)(kp + 32 + 16 * lh);
      kf.h[1] = *(const v8h*)(kp + 32 + 16 * lh + 8);
      sa = wmma_f16(qf[1].v, kf.v, sa);
#pragma unroll
      for (int r = 0; r < 8; ++r) s1[r] = sa[r] * 0.125f;
    }

    // ---- causal mask ----
    const int key0 = k0 + lnn;
    const int key1 = k0 + 16 + lnn;
#pragma unroll
    for (int r = 0; r < 8; ++r) {
      const int qrow = qbase + r + 8 * lh;
      if (key0 > qrow) s0[r] = -1e30f;
      if (key1 > qrow) s1[r] = -1e30f;
    }

    // ---- online softmax update (DPP reductions across 16-lane halves) ----
    float p0[8], p1[8];
#pragma unroll
    for (int r = 0; r < 8; ++r) {
      const float tmax = red_max16(fmaxf(s0[r], s1[r]));
      const float mnew = fmaxf(mrow[r], tmax);
      const float corr = __expf(mrow[r] - mnew);
      p0[r] = __expf(s0[r] - mnew);
      p1[r] = __expf(s1[r] - mnew);
      const float rsum = red_add16(p0[r] + p1[r]);
      lrow[r] = lrow[r] * corr + rsum;
      mrow[r] = mnew;
#pragma unroll
      for (int nc = 0; nc < 4; ++nc) o[nc][r] *= corr;
    }

    // ---- P: C-layout -> A-layout via per-wave LDS staging ----
#pragma unroll
    for (int r = 0; r < 8; ++r) {
      pbuf[wv][r + 8 * lh][lnn]      = (_Float16)p0[r];
      pbuf[wv][r + 8 * lh][16 + lnn] = (_Float16)p1[r];
    }
    __builtin_amdgcn_wave_barrier();
    asm volatile("s_wait_dscnt 0" ::: "memory");
    U16 pa;
    pa.h[0] = *(const v8h*)&pbuf[wv][lnn][16 * lh];
    pa.h[1] = *(const v8h*)&pbuf[wv][lnn][16 * lh + 8];
    __builtin_amdgcn_wave_barrier();

    // ---- O += P V  (V^T layout gives contiguous key runs) ----
#pragma unroll
    for (int nc = 0; nc < 4; ++nc) {
      const _Float16* vp =
          vT + ((size_t)b * 1024 + h * 64 + nc * 16 + lnn) * 2048 + k0 + 16 * lh;
      U16 vf;
      vf.h[0] = *(const v8h*)(vp);
      vf.h[1] = *(const v8h*)(vp + 8);
      o[nc] = wmma_f16(pa.v, vf.v, o[nc]);
    }
  }

  // ---- normalize and store f16 ----
#pragma unroll
  for (int nc = 0; nc < 4; ++nc) {
#pragma unroll
    for (int r = 0; r < 8; ++r) {
      const float val = o[nc][r] / lrow[r];
      const int m = r + 8 * lh;
      yout[(rowbase + qbase + m) * 1024 + h * 64 + nc * 16 + lnn] = (_Float16)val;
    }
  }
}

// ---------------------------------------------------------------------------
// Host orchestration
// ---------------------------------------------------------------------------
extern "C" void kernel_launch(void* const* d_in, const int* in_sizes, int n_in,
                              void* d_out, int out_size, void* d_ws, size_t ws_size,
                              hipStream_t stream) {
  (void)in_sizes; (void)n_in; (void)out_size; (void)ws_size;

  const float* x      = (const float*)d_in[0];
  const float* ln1_w  = (const float*)d_in[1];
  const float* ln1_b  = (const float*)d_in[2];
  const float* W_attn = (const float*)d_in[3];
  const float* b_attn = (const float*)d_in[4];
  const float* W_proj = (const float*)d_in[5];
  const float* b_proj = (const float*)d_in[6];
  const float* ln2_w  = (const float*)d_in[7];
  const float* ln2_b  = (const float*)d_in[8];
  const float* W_fc   = (const float*)d_in[9];
  const float* b_fc   = (const float*)d_in[10];
  const float* W_out  = (const float*)d_in[11];
  const float* b_out  = (const float*)d_in[12];
  float* out = (float*)d_out;

  const int M = 8192;          // B*T
  char* ws = (char*)d_ws;
  size_t off = 0;
  auto alloc = [&](size_t bytes) {
    size_t o = off;
    off += (bytes + 255) & ~(size_t)255;
    return (void*)(ws + o);
  };

  _Float16* wattn16 = (_Float16*)alloc((size_t)3072 * 1024 * 2);
  _Float16* wproj16 = (_Float16*)alloc((size_t)1024 * 1024 * 2);
  _Float16* wfc16   = (_Float16*)alloc((size_t)4096 * 1024 * 2);
  _Float16* wout16  = (_Float16*)alloc((size_t)1024 * 4096 * 2);
  _Float16* h16     = (_Float16*)alloc((size_t)M * 1024 * 2);
  _Float16* qkv16   = (_Float16*)alloc((size_t)M * 3072 * 2);
  _Float16* vT16    = (_Float16*)alloc((size_t)4 * 1024 * 2048 * 2);
  _Float16* attn16  = (_Float16*)alloc((size_t)M * 1024 * 2);
  float*    x1      = (float*)   alloc((size_t)M * 1024 * 4);
  _Float16* fc16    = (_Float16*)alloc((size_t)M * 4096 * 2);

  // 1. weight conversion to f16
  auto cvt = [&](const float* s, _Float16* d, int n) {
    cvt_kernel<<<(n + 255) / 256, 256, 0, stream>>>(s, d, n);
  };
  cvt(W_attn, wattn16, 3072 * 1024);
  cvt(W_proj, wproj16, 1024 * 1024);
  cvt(W_fc,   wfc16,   4096 * 1024);
  cvt(W_out,  wout16,  1024 * 4096);

  // 2. LN1
  ln_kernel<<<M, 256, 0, stream>>>(x, ln1_w, ln1_b, h16);

  // 3. QKV GEMM: [8192,1024] x [3072,1024]^T -> f16
  gemm_kernel<false, false, true><<<dim3(3072 / 256, M / 128), 256, 0, stream>>>(
      h16, wattn16, b_attn, nullptr, qkv16, nullptr, M, 3072, 1024);

  // 4. transpose V slice
  vtrans_kernel<<<(4 * 2048 * 1024) / 256, 256, 0, stream>>>(qkv16, vT16);

  // 5. causal flash attention
  attn_kernel<<<2048, 128, 0, stream>>>(qkv16, vT16, attn16);

  // 6. proj GEMM + residual(x) -> x1 (f32)
  gemm_kernel<false, true, false><<<dim3(1024 / 256, M / 128), 256, 0, stream>>>(
      attn16, wproj16, b_proj, x, nullptr, x1, M, 1024, 1024);

  // 7. LN2
  ln_kernel<<<M, 256, 0, stream>>>(x1, ln2_w, ln2_b, h16);

  // 8. fc GEMM + GELU -> f16
  gemm_kernel<true, false, true><<<dim3(4096 / 256, M / 128), 256, 0, stream>>>(
      h16, wfc16, b_fc, nullptr, fc16, nullptr, M, 4096, 1024);

  // 9. out GEMM + residual(x1) -> d_out (f32)
  gemm_kernel<false, true, false><<<dim3(1024 / 256, M / 128), 256, 0, stream>>>(
      fc16, wout16, b_out, x1, nullptr, out, M, 1024, 4096);
}
